// MMOE_13907104105036
// MI455X (gfx1250) — matrix operand order repaired and checked by reading, not compiled, not run on hardware
//
#include <hip/hip_runtime.h>
#include <hip/hip_bf16.h>
#include <math.h>

#define BATCH 16384
#define NIN   512
#define NEXP  8
#define HEXP  256
#define NTASK 2
#define HTOW  64

#define WAVES 4      // waves per block; each wave owns 16 batch rows
#define BPAD  520    // B tile row stride in bf16 (1040B = 1024 + 16B pad) -> conflict-free b128 reads
#define NTILE (HEXP / 16 * NEXP)   // 128 (n,e) tiles

typedef __bf16 bf16;
typedef bf16  v16bf __attribute__((ext_vector_type(16)));
typedef bf16  v8bf  __attribute__((ext_vector_type(8)));
typedef float v8f   __attribute__((ext_vector_type(8)));
typedef unsigned int v4u __attribute__((ext_vector_type(4)));
typedef int  v4i __attribute__((ext_vector_type(4)));
typedef int  v8i __attribute__((ext_vector_type(8)));

union AFrag { v16bf v; v8bf h[2]; };

#if __has_builtin(__builtin_amdgcn_tensor_load_to_lds)
#define HAS_TDM 1
#else
#define HAS_TDM 0
#endif

#define WMMA_BF16(A, B, C) \
  __builtin_amdgcn_wmma_f32_16x16x32_bf16(false, (A), false, (B), (short)0, (C), false, false)

// ---------------- prep kernels ----------------

__global__ void k_cvt_x(const float* __restrict__ x, bf16* __restrict__ xb, int n) {
  int i = blockIdx.x * blockDim.x + threadIdx.x;
  if (i < n) xb[i] = (bf16)x[i];
}

// WeT[e][h][i] = We[e][i][h]
__global__ void k_tr_we(const float* __restrict__ We, bf16* __restrict__ WeT) {
  int idx = blockIdx.x * blockDim.x + threadIdx.x;
  if (idx >= NEXP * HEXP * NIN) return;
  int i = idx % NIN; int r = idx / NIN; int h = r % HEXP; int e = r / HEXP;
  WeT[idx] = (bf16)We[((size_t)e * NIN + i) * HEXP + h];
}

// WtT[t][o][h] = Wt[t][h][o]
__global__ void k_tr_wt(const float* __restrict__ Wt, bf16* __restrict__ WtT) {
  int idx = blockIdx.x * blockDim.x + threadIdx.x;
  if (idx >= NTASK * HTOW * HEXP) return;
  int h = idx % HEXP; int r = idx / HEXP; int o = r % HTOW; int t = r / HTOW;
  WtT[idx] = (bf16)Wt[((size_t)t * HEXP + h) * HTOW + o];
}

// gates[t][b][e] = softmax_e(x[b] . Wg[t][:,e] + bg[t][e])
__global__ void k_gates(const float* __restrict__ x, const float* __restrict__ Wg,
                        const float* __restrict__ bg, float* __restrict__ gates) {
  int row = blockIdx.x * blockDim.x + threadIdx.x;
  if (row >= BATCH) return;
  float acc[NTASK * NEXP];
  #pragma unroll
  for (int j = 0; j < NTASK * NEXP; ++j) acc[j] = 0.f;
  const float* xr = x + (size_t)row * NIN;
  for (int i = 0; i < NIN; ++i) {
    float xv = xr[i];
    #pragma unroll
    for (int t = 0; t < NTASK; ++t)
      #pragma unroll
      for (int e = 0; e < NEXP; ++e)
        acc[t * NEXP + e] = fmaf(xv, Wg[((size_t)t * NIN + i) * NEXP + e], acc[t * NEXP + e]);
  }
  #pragma unroll
  for (int t = 0; t < NTASK; ++t) {
    float mx = -1e30f;
    #pragma unroll
    for (int e = 0; e < NEXP; ++e) { acc[t*NEXP+e] += bg[t*NEXP+e]; mx = fmaxf(mx, acc[t*NEXP+e]); }
    float ex[NEXP], s = 0.f;
    #pragma unroll
    for (int e = 0; e < NEXP; ++e) { ex[e] = __expf(acc[t*NEXP+e] - mx); s += ex[e]; }
    float inv = 1.f / s;
    #pragma unroll
    for (int e = 0; e < NEXP; ++e) gates[((size_t)t * BATCH + row) * NEXP + e] = ex[e] * inv;
  }
}

// ---------------- TDM staging: one 16x512 bf16 B-tile -> LDS (padded rows) ----------------

__device__ __forceinline__ void tdm_issue(unsigned lds_off, const bf16* gsrc) {
#if HAS_TDM
  unsigned long long ga = (unsigned long long)(size_t)gsrc;
  v4u g0;
  g0[0] = 1u;                                   // count=1, user mode
  g0[1] = lds_off;                              // lds_addr (bytes)
  g0[2] = (unsigned)(ga & 0xFFFFFFFFu);         // global_addr lo
  g0[3] = (unsigned)(ga >> 32) | 0x80000000u;   // global_addr hi | type=2 ("image")
  v8i g1;
  g1[0] = (1 << 16) | (1 << 20) | (7 << 22) | (3 << 25); // data_size=2B, pad_en, intvl=256dw, pad=4dw
  g1[1] = (int)(512u << 16);                    // tensor_dim0 = 512
  g1[2] = (int)(16u  << 16);                    // tensor_dim1 = 16
  g1[3] = (int)(512u << 16);                    // tile_dim0 = 512
  g1[4] = 16;                                   // tile_dim1 = 16, tile_dim2 = 0
  g1[5] = 512;                                  // tensor_dim0_stride = 512
  g1[6] = 0;
  g1[7] = 0;
  v4i z4 = {0, 0, 0, 0};
#if __clang_major__ >= 23
  v8i z8 = {0, 0, 0, 0, 0, 0, 0, 0};
  __builtin_amdgcn_tensor_load_to_lds(g0, g1, z4, z4, z8, 0);
#else
  __builtin_amdgcn_tensor_load_to_lds(g0, g1, z4, z4, 0);
#endif
#else
  (void)lds_off; (void)gsrc;
#endif
}

// ---------------- phase 1: experts + gated mixture -> tower_in (bf16) ----------------

__global__ __launch_bounds__(WAVES * 32) void k_mmoe(
    const bf16* __restrict__ Xb, const bf16* __restrict__ WeT,
    const float* __restrict__ be, const float* __restrict__ gates,
    bf16* __restrict__ tin)   // [T][B][HEXP]
{
  __shared__ __align__(16) bf16 Bb[3][16][BPAD];   // 49,920 B triple-buffered TDM dest

  const int tid  = threadIdx.x;
  const int wave = tid >> 5;
  const int lane = tid & 31;
  const int col  = lane & 15;
  const int hi   = lane >> 4;
  const int rowBase = (blockIdx.x * WAVES + wave) * 16;

  const unsigned bbOff =
      (unsigned)(size_t)(__attribute__((address_space(3))) void*)&Bb[0][0][0];
  const unsigned bbTile = 16u * BPAD * sizeof(bf16);

  // preload A fragments (16 rows x K=512) into 128 VGPRs
  v16bf aF[16];
  {
    const bf16* xr = Xb + (size_t)(rowBase + col) * NIN;
    #pragma unroll
    for (int k = 0; k < 16; ++k) {
      AFrag a;
      a.h[0] = *(const v8bf*)(xr + k * 32 + hi * 16);
      a.h[1] = *(const v8bf*)(xr + k * 32 + hi * 16 + 8);
      aF[k] = a.v;
    }
  }

  const float* g0p = gates + (size_t)rowBase * NEXP;
  const float* g1p = gates + ((size_t)BATCH + rowBase) * NEXP;

  // prologue: stage tiles 0 and 1
  if (wave == 0) {
#if HAS_TDM
    tdm_issue(bbOff + 0 * bbTile, WeT);                      // tile 0 (n=0,e=0)
    tdm_issue(bbOff + 1 * bbTile, WeT + (size_t)HEXP * NIN); // tile 1 (n=0,e=1)
#else
    #pragma unroll 2
    for (int tI = 0; tI < 2; ++tI) {
      const bf16* src = WeT + (size_t)tI * HEXP * NIN;
      for (int j = 0; j < 16; ++j) {
        *(v8bf*)&Bb[tI][j][lane * 16 + 0] = *(const v8bf*)(src + (size_t)j * NIN + lane * 16);
        *(v8bf*)&Bb[tI][j][lane * 16 + 8] = *(const v8bf*)(src + (size_t)j * NIN + lane * 16 + 8);
      }
    }
#endif
  }

  int cur = 0;   // current LDS buffer (= idx % 3)
  for (int n = 0; n < HEXP / 16; ++n) {
    v8f t0 = {}; v8f t1 = {};
    for (int e = 0; e < NEXP; ++e) {
      const int idx = n * NEXP + e;
#if HAS_TDM
      if (wave == 0) {
        if (idx == NTILE - 1) __builtin_amdgcn_s_wait_tensorcnt(0);
        else                  __builtin_amdgcn_s_wait_tensorcnt(1);  // tile idx complete (in-order)
      }
#endif
      __syncthreads();   // tile idx visible; everyone done with tile idx-1's buffer

      // stage tile idx+2 into the buffer last read at idx-1 (safe after barrier)
      if (wave == 0 && idx + 2 < NTILE) {
        const int nx = idx + 2;
        const int tgt = (cur == 0) ? 2 : cur - 1;          // (idx+2) % 3
        const bf16* src = WeT + ((size_t)((nx & 7) * HEXP + (nx >> 3) * 16)) * NIN;
#if HAS_TDM
        tdm_issue(bbOff + (unsigned)tgt * bbTile, src);
#else
        for (int j = 0; j < 16; ++j) {
          *(v8bf*)&Bb[tgt][j][lane * 16 + 0] = *(const v8bf*)(src + (size_t)j * NIN + lane * 16);
          *(v8bf*)&Bb[tgt][j][lane * 16 + 8] = *(const v8bf*)(src + (size_t)j * NIN + lane * 16 + 8);
        }
#endif
      }

      // 16 WMMAs over K=512: explicit 4-slot B circular buffer (forces ~8 ds ops
      // in flight -> partial dscnt waits) + two interleaved accumulator chains.
      const bf16* bp = &Bb[cur][col][hi * 16];
#define LDB(kk) (*(const v16bf*)(bp + (kk) * 32))
      v16bf B0 = LDB(0), B1 = LDB(1), B2 = LDB(2), B3 = LDB(3);
      v8f accA = {}, accB = {};
      accA = WMMA_BF16(aF[0],  B0, accA); B0 = LDB(4);
      accB = WMMA_BF16(aF[1],  B1, accB); B1 = LDB(5);
      accA = WMMA_BF16(aF[2],  B2, accA); B2 = LDB(6);
      accB = WMMA_BF16(aF[3],  B3, accB); B3 = LDB(7);
      accA = WMMA_BF16(aF[4],  B0, accA); B0 = LDB(8);
      accB = WMMA_BF16(aF[5],  B1, accB); B1 = LDB(9);
      accA = WMMA_BF16(aF[6],  B2, accA); B2 = LDB(10);
      accB = WMMA_BF16(aF[7],  B3, accB); B3 = LDB(11);
      accA = WMMA_BF16(aF[8],  B0, accA); B0 = LDB(12);
      accB = WMMA_BF16(aF[9],  B1, accB); B1 = LDB(13);
      accA = WMMA_BF16(aF[10], B2, accA); B2 = LDB(14);
      accB = WMMA_BF16(aF[11], B3, accB); B3 = LDB(15);
      accA = WMMA_BF16(aF[12], B0, accA);
      accB = WMMA_BF16(aF[13], B1, accB);
      accA = WMMA_BF16(aF[14], B2, accA);
      accB = WMMA_BF16(aF[15], B3, accB);
#undef LDB

      float bias = be[e * HEXP + n * 16 + col];
      #pragma unroll
      for (int r = 0; r < 8; ++r) {
        float v = fmaxf(accA[r] + accB[r] + bias, 0.f);
        int ro = (r + hi * 8) * NEXP + e;
        t0[r] = fmaf(g0p[ro], v, t0[r]);
        t1[r] = fmaf(g1p[ro], v, t1[r]);
      }
      cur = (cur == 2) ? 0 : cur + 1;
    }
    // write tower_in (bf16) for this 16-col H tile
    #pragma unroll
    for (int r = 0; r < 8; ++r) {
      size_t row = (size_t)rowBase + r + hi * 8;
      tin[(0 * (size_t)BATCH + row) * HEXP + n * 16 + col] = (bf16)t0[r];
      tin[(1 * (size_t)BATCH + row) * HEXP + n * 16 + col] = (bf16)t1[r];
    }
  }
}

// ---------------- phase 2: towers (K=256 WMMA) + sigmoid ----------------

__global__ __launch_bounds__(WAVES * 32) void k_tower(
    const bf16* __restrict__ tin, const bf16* __restrict__ WtT,
    const float* __restrict__ bt, float* __restrict__ out)
{
  const int tid  = threadIdx.x;
  const int wave = tid >> 5;
  const int lane = tid & 31;
  const int col  = lane & 15;
  const int hi   = lane >> 4;
  const int rowBase = (blockIdx.x * WAVES + wave) * 16;

  #pragma unroll
  for (int t = 0; t < NTASK; ++t) {
    v8f acc[HTOW / 16];
    #pragma unroll
    for (int ot = 0; ot < HTOW / 16; ++ot) acc[ot] = (v8f){};

    const bf16* ar = tin + ((size_t)t * BATCH + rowBase + col) * HEXP;
    #pragma unroll
    for (int k = 0; k < HEXP / 32; ++k) {
      AFrag a;
      a.h[0] = *(const v8bf*)(ar + k * 32 + hi * 16);
      a.h[1] = *(const v8bf*)(ar + k * 32 + hi * 16 + 8);
      #pragma unroll
      for (int ot = 0; ot < HTOW / 16; ++ot) {
        v16bf bF = *(const v16bf*)(WtT + ((size_t)(t * HTOW + ot * 16 + col)) * HEXP + k * 32 + hi * 16);
        acc[ot] = WMMA_BF16(a.v, bF, acc[ot]);
      }
    }
    #pragma unroll
    for (int ot = 0; ot < HTOW / 16; ++ot) {
      float bias = bt[t * HTOW + ot * 16 + col];
      #pragma unroll
      for (int r = 0; r < 8; ++r) {
        float v = acc[ot][r] + bias;
        float s = 1.f / (1.f + __expf(-v));
        out[((size_t)t * BATCH + rowBase + r + hi * 8) * HTOW + ot * 16 + col] = s;
      }
    }
  }
}

// ---------------- launcher ----------------

extern "C" void kernel_launch(void* const* d_in, const int* in_sizes, int n_in,
                              void* d_out, int out_size, void* d_ws, size_t ws_size,
                              hipStream_t stream) {
  const float* xv = (const float*)d_in[0];
  const float* We = (const float*)d_in[1];
  const float* be = (const float*)d_in[2];
  const float* Wg = (const float*)d_in[3];
  const float* bg = (const float*)d_in[4];
  const float* Wt = (const float*)d_in[5];
  const float* bt = (const float*)d_in[6];
  float* out = (float*)d_out;

  char* ws = (char*)d_ws;
  size_t off = 0;
  auto carve = [&](size_t bytes) -> char* {
    char* p = ws + off;
    off += (bytes + 255) & ~(size_t)255;
    return p;
  };
  bf16*  Xb    = (bf16*) carve((size_t)BATCH * NIN * sizeof(bf16));
  bf16*  WeT   = (bf16*) carve((size_t)NEXP * HEXP * NIN * sizeof(bf16));
  bf16*  WtT   = (bf16*) carve((size_t)NTASK * HTOW * HEXP * sizeof(bf16));
  float* gates = (float*)carve((size_t)NTASK * BATCH * NEXP * sizeof(float));
  bf16*  tin   = (bf16*) carve((size_t)NTASK * BATCH * HEXP * sizeof(bf16));

  { int n = BATCH * NIN;         k_cvt_x<<<(n + 255) / 256, 256, 0, stream>>>(xv, Xb, n); }
  { int n = NEXP * HEXP * NIN;   k_tr_we<<<(n + 255) / 256, 256, 0, stream>>>(We, WeT); }
  { int n = NTASK * HTOW * HEXP; k_tr_wt<<<(n + 255) / 256, 256, 0, stream>>>(Wt, WtT); }
  k_gates<<<BATCH / 256, 256, 0, stream>>>(xv, Wg, bg, gates);
  k_mmoe <<<BATCH / (16 * WAVES), WAVES * 32, 0, stream>>>(Xb, WeT, be, gates, tin);
  k_tower<<<BATCH / (16 * WAVES), WAVES * 32, 0, stream>>>(tin, WtT, bt, out);
}
